// ROLANDGNN_7395933684286
// MI455X (gfx1250) — matrix operand order repaired and checked by reading, not compiled, not run on hardware
//
#include <hip/hip_runtime.h>
#include <hip/hip_bf16.h>
#include <stdint.h>

// ---------------------------------------------------------------------------
// CDNA5 (gfx1250) wave32 WMMA bf16 + TDM implementation of ROLAND GNN forward.
// ---------------------------------------------------------------------------

typedef __attribute__((ext_vector_type(16))) __bf16 v16bf;
typedef __attribute__((ext_vector_type(8)))  float  v8f;
typedef __attribute__((ext_vector_type(4)))  unsigned int u32x4;
typedef __attribute__((ext_vector_type(8)))  int i32x8;
typedef __attribute__((ext_vector_type(4)))  int i32x4;

__device__ __forceinline__ v8f wmma_bf16(v16bf a, v16bf b, v8f c) {
  return __builtin_amdgcn_wmma_f32_16x16x32_bf16(
      false, a, false, b, (short)0, c, false, false);
}

// A fragment: 16x32 bf16 tile, row-major, leading dim `ld`.
__device__ __forceinline__ v16bf a_frag(const __bf16* tile, int ld, int lane) {
  int m  = lane & 15;
  int kb = (lane >> 4) << 3;            // 0 or 8
  const __bf16* row = tile + m * ld;
  v16bf f;
#pragma unroll
  for (int j = 0; j < 8; ++j) {
    int k = (j < 4) ? (kb + 2 * j) : (kb + 16 + 2 * (j - 4));
    f[2 * j]     = row[k];
    f[2 * j + 1] = row[k + 1];
  }
  return f;
}

// B = W^T tile (32x16): W row-major [nout, K] bf16, B[k][n] = W[n0+n][k0+k].
__device__ __forceinline__ v16bf b_frag(const __bf16* W, int ldw,
                                        int n0, int k0, int lane) {
  int n  = lane & 15;
  int kb = (lane >> 4) << 3;            // 0 or 8
  const __bf16* row = W + (size_t)(n0 + n) * ldw + k0 + kb;
  v16bf f;
#pragma unroll
  for (int j = 0; j < 8; ++j) {
    int k = (j < 4) ? (2 * j) : (16 + 2 * (j - 4));
    f[2 * j]     = row[k];
    f[2 * j + 1] = row[k + 1];
  }
  return f;
}

__device__ __forceinline__ float leaky(float v) {
  return v >= 0.f ? v : 0.01f * v;
}
__device__ __forceinline__ float sigm(float v) {
  return 1.f / (1.f + __expf(-v));
}

// Stage ROWS x K f32 rows (zero-padded past M) into LDS as bf16 (VALU path).
template <int ROWS, int K>
__device__ __forceinline__ void stage_bf16(const float* __restrict__ A, int M,
                                           int row0, __bf16* sA) {
  const int total = ROWS * K;
  for (int idx = threadIdx.x * 4; idx < total; idx += blockDim.x * 4) {
    int r = idx / K, c = idx % K;
    int gr = row0 + r;
    float4 v = make_float4(0.f, 0.f, 0.f, 0.f);
    if (gr < M) v = *(const float4*)(A + (size_t)gr * K + c);
    sA[idx + 0] = (__bf16)v.x;
    sA[idx + 1] = (__bf16)v.y;
    sA[idx + 2] = (__bf16)v.z;
    sA[idx + 3] = (__bf16)v.w;
  }
}

// ---------------------------------------------------------------------------
// TDM stage of a 64x128 bf16 tile (row-major, ld=128) straight into LDS.
// Rows past M are zero-filled by TDM OOB semantics (tensor_dim1 = M-row0).
// Issue from ONE wave only; follow with s_wait_tensorcnt + block barrier.
// ---------------------------------------------------------------------------
__device__ __forceinline__ void tdm_stage_64x128(const __bf16* gbase, int M,
                                                 int row0, __bf16* sdst) {
#if __has_builtin(__builtin_amdgcn_tensor_load_to_lds)
  uint64_t ga  = (uint64_t)(uintptr_t)(gbase + (size_t)row0 * 128);
  uint32_t lds = (uint32_t)(uintptr_t)sdst;   // LDS aperture: offset in addr[31:0]
  int rem = M - row0;
  if (rem > 64) rem = 64;
  if (rem < 0)  rem = 0;

  u32x4 g0;
  g0[0] = 1u;                                   // count=1, user descriptor
  g0[1] = lds;                                  // lds_addr
  g0[2] = (uint32_t)ga;                         // global_addr[31:0]
  g0[3] = (uint32_t)((ga >> 32) & 0x01FFFFFFu)  // global_addr[56:32]
          | (2u << 30);                         // type=2 (image)

  i32x8 g1;
  g1[0] = (1 << 16);          // data_size=1 (2 bytes); mask/pad/iterate = 0
  g1[1] = (128 << 16);        // tensor_dim0 = 128 (bits 79:48)
  g1[2] = (rem << 16);        // tensor_dim1 = remaining rows (bits 111:80)
  g1[3] = (128 << 16);        // tile_dim0 = 128 (bits 127:112)
  g1[4] = 64;                 // tile_dim1 = 64 rows (bits 143:128)
  g1[5] = 128;                // tensor_dim0_stride = 128 (bits 207:160)
  g1[6] = 0;
  g1[7] = 0;

  i32x4 g2 = {0, 0, 0, 0};    // 2D tensor: groups 2/3 unused
  i32x4 g3 = {0, 0, 0, 0};
  i32x8 g4 = {0, 0, 0, 0, 0, 0, 0, 0};  // extra group (clang-23 6-arg form)
  __builtin_amdgcn_tensor_load_to_lds(g0, g1, g2, g3, g4, 0);
  __builtin_amdgcn_s_wait_tensorcnt(0);
#else
  // fallback: cooperative vector copy (wave0 only, 32 lanes)
  for (int idx = (int)(threadIdx.x & 31) * 8; idx < 64 * 128; idx += 32 * 8) {
    int r = idx >> 7, c = idx & 127;
    int gr = row0 + r;
    uint4 v = make_uint4(0, 0, 0, 0);
    if (gr < M) v = *(const uint4*)(gbase + (size_t)gr * 128 + c);
    *(uint4*)(sdst + idx) = v;
  }
#endif
}

// ---------------------------------------------------------------------------
// f32 -> bf16 conversion
// ---------------------------------------------------------------------------
__global__ void cvt_bf16_kernel(const float* __restrict__ src,
                                __bf16* __restrict__ dst, int n) {
  int i = blockIdx.x * blockDim.x + threadIdx.x;
  if (i < n) dst[i] = (__bf16)src[i];
}

// ---------------------------------------------------------------------------
// Fused preprocess: outb = bf16(leaky(leaky(x@W1^T+b1)@W2^T+b2))
// x is f32 (external input) -> VALU staging; h1 lives only in LDS.
// ---------------------------------------------------------------------------
__global__ __launch_bounds__(256)
void pre_kernel(const float* __restrict__ x,
                const __bf16* __restrict__ W1, const float* __restrict__ b1,
                const __bf16* __restrict__ W2, const float* __restrict__ b2,
                __bf16* __restrict__ outb, int M) {
  __shared__ __bf16 sA[64 * 128];   // 16 KB
  __shared__ __bf16 sH[64 * 256];   // 32 KB
  int row0 = blockIdx.x * 64;
  stage_bf16<64, 128>(x, M, row0, sA);
  __syncthreads();

  int lane = threadIdx.x & 31;
  int wave = threadIdx.x >> 5;
  int n    = lane & 15;
  int mhi  = (lane >> 4) << 3;

  for (int t = wave; t < 64; t += 8) {           // h1 = leaky(x@W1^T+b1)
    int mt = t & 3, nt = t >> 2;
    v8f acc = {};
#pragma unroll
    for (int ks = 0; ks < 4; ++ks)
      acc = wmma_bf16(a_frag(sA + mt * 16 * 128 + ks * 32, 128, lane),
                      b_frag(W1, 128, nt * 16, ks * 32, lane), acc);
    float bias = b1[nt * 16 + n];
#pragma unroll
    for (int r = 0; r < 8; ++r)
      sH[(mt * 16 + r + mhi) * 256 + nt * 16 + n] = (__bf16)leaky(acc[r] + bias);
  }
  __syncthreads();

  for (int t = wave; t < 32; t += 8) {           // out = leaky(h1@W2^T+b2)
    int mt = t & 3, nt = t >> 2;
    v8f acc = {};
#pragma unroll
    for (int ks = 0; ks < 8; ++ks)
      acc = wmma_bf16(a_frag(sH + mt * 16 * 256 + ks * 32, 256, lane),
                      b_frag(W2, 256, nt * 16, ks * 32, lane), acc);
    float bias = b2[nt * 16 + n];
#pragma unroll
    for (int r = 0; r < 8; ++r) {
      int row = row0 + mt * 16 + r + mhi;
      if (row < M)
        outb[(size_t)row * 128 + nt * 16 + n] = (__bf16)leaky(acc[r] + bias);
    }
  }
}

// ---------------------------------------------------------------------------
// xw = A @ W^T (K=128, Nout=128). A is bf16 in global, staged by TDM.
// ---------------------------------------------------------------------------
__global__ __launch_bounds__(256)
void xw_kernel(const __bf16* __restrict__ Ab, const __bf16* __restrict__ W,
               float* __restrict__ out, int M) {
  __shared__ __bf16 sA[64 * 128];
  int row0 = blockIdx.x * 64;
  if (threadIdx.x < 32) tdm_stage_64x128(Ab, M, row0, sA);
  __syncthreads();

  int lane = threadIdx.x & 31, wave = threadIdx.x >> 5;
  int n = lane & 15, mhi = (lane >> 4) << 3;
  for (int t = wave; t < 32; t += 8) {
    int mt = t & 3, nt = t >> 2;
    v8f acc = {};
#pragma unroll
    for (int ks = 0; ks < 4; ++ks)
      acc = wmma_bf16(a_frag(sA + mt * 16 * 128 + ks * 32, 128, lane),
                      b_frag(W, 128, nt * 16, ks * 32, lane), acc);
#pragma unroll
    for (int r = 0; r < 8; ++r) {
      int row = row0 + mt * 16 + r + mhi;
      if (row < M) out[(size_t)row * 128 + nt * 16 + n] = acc[r];
    }
  }
}

// ---------------------------------------------------------------------------
// Fused GRU cell.  x-path input is bf16 (TDM staged); hprev is f32 (external)
// staged via VALU convert.  Writes f32 embedding + bf16 copy for next GEMM.
// ---------------------------------------------------------------------------
__global__ __launch_bounds__(256)
void gru_kernel(const __bf16* __restrict__ xhb, const float* __restrict__ hprev,
                const __bf16* __restrict__ Wih, const __bf16* __restrict__ Whh,
                const float* __restrict__ bih, const float* __restrict__ bhh,
                float* __restrict__ outf, __bf16* __restrict__ outb, int M) {
  __shared__ __bf16 sX[64 * 128];   // 16 KB
  __shared__ __bf16 sH[64 * 128];   // 16 KB
  int row0 = blockIdx.x * 64;
  if (threadIdx.x < 32) tdm_stage_64x128(xhb, M, row0, sX);
  stage_bf16<64, 128>(hprev, M, row0, sH);
  __syncthreads();

  int lane = threadIdx.x & 31, wave = threadIdx.x >> 5;
  int n = lane & 15, mhi = (lane >> 4) << 3;

  for (int t = wave; t < 32; t += 8) {
    int mt = t & 3, ct = t >> 2;
    int c = ct * 16 + n;
    v8f air = {}, aiz = {}, ain = {}, ahr = {}, ahz = {}, ahn = {};
#pragma unroll
    for (int ks = 0; ks < 4; ++ks) {
      v16bf ax = a_frag(sX + mt * 16 * 128 + ks * 32, 128, lane);
      v16bf ah = a_frag(sH + mt * 16 * 128 + ks * 32, 128, lane);
      air = wmma_bf16(ax, b_frag(Wih, 128,       ct * 16, ks * 32, lane), air);
      aiz = wmma_bf16(ax, b_frag(Wih, 128, 128 + ct * 16, ks * 32, lane), aiz);
      ain = wmma_bf16(ax, b_frag(Wih, 128, 256 + ct * 16, ks * 32, lane), ain);
      ahr = wmma_bf16(ah, b_frag(Whh, 128,       ct * 16, ks * 32, lane), ahr);
      ahz = wmma_bf16(ah, b_frag(Whh, 128, 128 + ct * 16, ks * 32, lane), ahz);
      ahn = wmma_bf16(ah, b_frag(Whh, 128, 256 + ct * 16, ks * 32, lane), ahn);
    }
    float bir = bih[c],       bhr = bhh[c];
    float biz = bih[128 + c], bhz = bhh[128 + c];
    float bin = bih[256 + c], bhn = bhh[256 + c];
#pragma unroll
    for (int r = 0; r < 8; ++r) {
      int row = row0 + mt * 16 + r + mhi;
      if (row < M) {
        float rg = sigm(air[r] + bir + ahr[r] + bhr);
        float zg = sigm(aiz[r] + biz + ahz[r] + bhz);
        float ng = tanhf(ain[r] + bin + rg * (ahn[r] + bhn));
        float hp = hprev[(size_t)row * 128 + c];
        float h  = (1.f - zg) * ng + zg * hp;
        outf[(size_t)row * 128 + c] = h;
        outb[(size_t)row * 128 + c] = (__bf16)h;
      }
    }
  }
}

// ---------------------------------------------------------------------------
// GCN aggregation helpers
// ---------------------------------------------------------------------------
__global__ void deg_init_kernel(float* __restrict__ deg, int N) {
  int i = blockIdx.x * blockDim.x + threadIdx.x;
  if (i < N) deg[i] = 1.0f;   // self-loop
}
__global__ void deg_count_kernel(const int* __restrict__ dst,
                                 float* __restrict__ deg, int E) {
  int e = blockIdx.x * blockDim.x + threadIdx.x;
  if (e < E) atomicAdd(&deg[dst[e]], 1.0f);
}
__global__ void dinv_kernel(const float* __restrict__ deg,
                            float* __restrict__ dinv, int N) {
  int i = blockIdx.x * blockDim.x + threadIdx.x;
  if (i < N) { float d = deg[i]; dinv[i] = d > 0.f ? rsqrtf(d) : 0.f; }
}
__global__ void agg_init_kernel(const float* __restrict__ xw,
                                const float* __restrict__ dinv,
                                float* __restrict__ agg, int N) {
  int i = blockIdx.x * blockDim.x + threadIdx.x;
  if (i < N * 32) {
    int node = i >> 5, c4 = (i & 31) * 4;
    float w = dinv[node]; w *= w;
    float4 v = *(const float4*)(xw + (size_t)node * 128 + c4);
    v.x *= w; v.y *= w; v.z *= w; v.w *= w;
    *(float4*)(agg + (size_t)node * 128 + c4) = v;
  }
}
__global__ void scatter_kernel(const int* __restrict__ src,
                               const int* __restrict__ dst,
                               const float* __restrict__ xw,
                               const float* __restrict__ dinv,
                               float* __restrict__ agg, int E) {
  int gw = (blockIdx.x * blockDim.x + threadIdx.x) >> 5;
  int lane = threadIdx.x & 31;
  if (gw >= E) return;
  int s = src[gw], d = dst[gw];
  float w = dinv[s] * dinv[d];
  float4 v = *(const float4*)(xw + (size_t)s * 128 + lane * 4);
  float* o = agg + (size_t)d * 128 + lane * 4;
  atomicAdd(o + 0, v.x * w);
  atomicAdd(o + 1, v.y * w);
  atomicAdd(o + 2, v.z * w);
  atomicAdd(o + 3, v.w * w);
}
__global__ void bias_leaky_kernel(const float* __restrict__ agg,
                                  const float* __restrict__ b,
                                  __bf16* __restrict__ outb, int N) {
  int i = blockIdx.x * blockDim.x + threadIdx.x;
  if (i < N * 32) {
    int node = i >> 5, c4 = (i & 31) * 4;
    float4 v = *(const float4*)(agg + (size_t)node * 128 + c4);
    float4 bb = *(const float4*)(b + c4);
    __bf16* o = outb + (size_t)node * 128 + c4;
    o[0] = (__bf16)leaky(v.x + bb.x);
    o[1] = (__bf16)leaky(v.y + bb.y);
    o[2] = (__bf16)leaky(v.z + bb.z);
    o[3] = (__bf16)leaky(v.w + bb.w);
  }
}

// ---------------------------------------------------------------------------
// LP head
// ---------------------------------------------------------------------------
__global__ void wsum_kernel(const float* __restrict__ w_post,
                            const float* __restrict__ b_post,
                            float* __restrict__ wsum) {
  int c = threadIdx.x;
  if (c < 128) wsum[c] = w_post[c] + w_post[128 + c];
  if (c == 0) wsum[128] = b_post[0] + b_post[1];
}
__global__ void head_kernel(const int* __restrict__ eli,
                            const float* __restrict__ emb,
                            const float* __restrict__ wsum,
                            float* __restrict__ logits, int EL) {
  int gw = (blockIdx.x * blockDim.x + threadIdx.x) >> 5;
  int lane = threadIdx.x & 31;
  if (gw >= EL) return;
  int s = eli[gw], d = eli[EL + gw];
  float4 a = *(const float4*)(emb + (size_t)s * 128 + lane * 4);
  float4 b = *(const float4*)(emb + (size_t)d * 128 + lane * 4);
  float4 w = *(const float4*)(wsum + lane * 4);
  float p = a.x * b.x * w.x + a.y * b.y * w.y +
            a.z * b.z * w.z + a.w * b.w * w.w;
#pragma unroll
  for (int off = 16; off >= 1; off >>= 1) p += __shfl_xor(p, off, 32);
  if (lane == 0) logits[gw] = p + wsum[128];
}

// ---------------------------------------------------------------------------
// Host launch
// ---------------------------------------------------------------------------
extern "C" void kernel_launch(void* const* d_in, const int* in_sizes, int n_in,
                              void* d_out, int out_size, void* d_ws,
                              size_t ws_size, hipStream_t stream) {
  (void)n_in; (void)out_size; (void)ws_size;
  const float* x      = (const float*)d_in[0];
  const int*   eidx   = (const int*)d_in[1];
  const int*   eli    = (const int*)d_in[2];
  const float* prev0  = (const float*)d_in[3];
  const float* prev1  = (const float*)d_in[4];
  const float* w_pre1 = (const float*)d_in[5];
  const float* b_pre1 = (const float*)d_in[6];
  const float* w_pre2 = (const float*)d_in[7];
  const float* b_pre2 = (const float*)d_in[8];
  const float* w_c1   = (const float*)d_in[9];
  const float* b_c1   = (const float*)d_in[10];
  const float* w_c2   = (const float*)d_in[11];
  const float* b_c2   = (const float*)d_in[12];
  const float* w_post = (const float*)d_in[13];
  const float* b_post = (const float*)d_in[14];
  const float* g1_wih = (const float*)d_in[15];
  const float* g1_whh = (const float*)d_in[16];
  const float* g1_bih = (const float*)d_in[17];
  const float* g1_bhh = (const float*)d_in[18];
  const float* g2_wih = (const float*)d_in[19];
  const float* g2_whh = (const float*)d_in[20];
  const float* g2_bih = (const float*)d_in[21];
  const float* g2_bhh = (const float*)d_in[22];

  const int N  = in_sizes[0] / 128;
  const int E  = in_sizes[1] / 2;
  const int EL = in_sizes[2] / 2;

  float* logits = (float*)d_out;
  float* emb0 = logits + EL;
  float* emb1 = emb0 + (size_t)N * 128;
  float* emb2 = emb1 + (size_t)N * 128;

  char* ws = (char*)d_ws;
  auto carve = [&](size_t bytes) {
    char* p = ws;
    ws += (bytes + 255) & ~(size_t)255;
    return p;
  };
  __bf16* W1b  = (__bf16*)carve(256 * 128 * 2);
  __bf16* W2b  = (__bf16*)carve(128 * 256 * 2);
  __bf16* Wc1b = (__bf16*)carve(128 * 128 * 2);
  __bf16* Wc2b = (__bf16*)carve(128 * 128 * 2);
  __bf16* G1ib = (__bf16*)carve(384 * 128 * 2);
  __bf16* G1hb = (__bf16*)carve(384 * 128 * 2);
  __bf16* G2ib = (__bf16*)carve(384 * 128 * 2);
  __bf16* G2hb = (__bf16*)carve(384 * 128 * 2);
  float*  bufA = (float*)carve((size_t)N * 128 * 4);   // xw output
  float*  bufC = (float*)carve((size_t)N * 128 * 4);   // aggregation
  __bf16* actb = (__bf16*)carve((size_t)N * 128 * 2);  // pre / bias_leaky out
  __bf16* embb = (__bf16*)carve((size_t)N * 128 * 2);  // bf16 embedding copy
  float*  deg  = (float*)carve((size_t)N * 4);
  float*  dinv = (float*)carve((size_t)N * 4);
  float*  wsum = (float*)carve(129 * 4);

  struct { const float* s; __bf16* d; int n; } cv[8] = {
    {w_pre1, W1b, 256 * 128}, {w_pre2, W2b, 128 * 256},
    {w_c1, Wc1b, 128 * 128},  {w_c2, Wc2b, 128 * 128},
    {g1_wih, G1ib, 384 * 128}, {g1_whh, G1hb, 384 * 128},
    {g2_wih, G2ib, 384 * 128}, {g2_whh, G2hb, 384 * 128},
  };
  for (int i = 0; i < 8; ++i)
    cvt_bf16_kernel<<<(cv[i].n + 255) / 256, 256, 0, stream>>>(cv[i].s, cv[i].d, cv[i].n);

  const int nblk = (N + 63) / 64;
  const int nv   = (N * 32 + 255) / 256;
  const int nb1  = (N + 255) / 256;
  const int eb   = (E + 255) / 256;
  const int ewv  = (E * 32 + 255) / 256;
  const int elwv = (EL * 32 + 255) / 256;

  // preprocess + GRU1 -> emb0
  pre_kernel<<<nblk, 256, 0, stream>>>(x, W1b, b_pre1, W2b, b_pre2, actb, N);
  gru_kernel<<<nblk, 256, 0, stream>>>(actb, prev0, G1ib, G1hb, g1_bih, g1_bhh,
                                       emb0, embb, N);

  // degrees (shared by both convs)
  deg_init_kernel<<<nb1, 256, 0, stream>>>(deg, N);
  deg_count_kernel<<<eb, 256, 0, stream>>>(eidx + E, deg, E);
  dinv_kernel<<<nb1, 256, 0, stream>>>(deg, dinv, N);

  // conv1 + GRU1 -> emb1
  xw_kernel<<<nblk, 256, 0, stream>>>(embb, Wc1b, bufA, N);
  agg_init_kernel<<<nv, 256, 0, stream>>>(bufA, dinv, bufC, N);
  scatter_kernel<<<ewv, 256, 0, stream>>>(eidx, eidx + E, bufA, dinv, bufC, E);
  bias_leaky_kernel<<<nv, 256, 0, stream>>>(bufC, b_c1, actb, N);
  gru_kernel<<<nblk, 256, 0, stream>>>(actb, prev0, G1ib, G1hb, g1_bih, g1_bhh,
                                       emb1, embb, N);

  // conv2 + GRU2 -> emb2
  xw_kernel<<<nblk, 256, 0, stream>>>(embb, Wc2b, bufA, N);
  agg_init_kernel<<<nv, 256, 0, stream>>>(bufA, dinv, bufC, N);
  scatter_kernel<<<ewv, 256, 0, stream>>>(eidx, eidx + E, bufA, dinv, bufC, E);
  bias_leaky_kernel<<<nv, 256, 0, stream>>>(bufC, b_c2, actb, N);
  gru_kernel<<<nblk, 256, 0, stream>>>(actb, prev1, G2ib, G2hb, g2_bih, g2_bhh,
                                       emb2, embb, N);

  // LP head
  wsum_kernel<<<1, 128, 0, stream>>>(w_post, b_post, wsum);
  head_kernel<<<elwv, 256, 0, stream>>>(eli, emb2, wsum, logits, EL);
}
